// DummyTransformer_75763223101889
// MI455X (gfx1250) — matrix operand order repaired and checked by reading, not compiled, not run on hardware
//
#include <hip/hip_runtime.h>
#include <math.h>

// ---------------------------------------------------------------------------
// Problem constants (reference: B=2, S=2048, E=1024, H=16, D=64, 2 shared layers)
// ---------------------------------------------------------------------------
constexpr int E  = 1024;
constexpr int H  = 16;
constexpr int D  = 64;
constexpr int B  = 2;
constexpr int S  = 2048;
constexpr int M  = B * S;      // 4096 GEMM rows
constexpr int N3 = 3 * E;      // 3072 packed-QKV cols
constexpr int NBH = B * H;     // 32 (batch*heads)
#define SCALE_QK 0.125f        // 1/sqrt(64)

typedef __attribute__((ext_vector_type(16))) __bf16 v16bf;
typedef __attribute__((ext_vector_type(8)))  float  v8f;

union FragU { v16bf v; uint4 q[2]; };

__device__ __forceinline__ v8f zero8() {
  v8f z = {0.f, 0.f, 0.f, 0.f, 0.f, 0.f, 0.f, 0.f};
  return z;
}

// A fragment 16x32 (MxK), A row-major with leading dim lda (bf16).
// Lane L: M = L&15 ; VGPR i holds K pair: K = (L>=16?8:0) + (i&3)*2 + (i>>2)*16
// => per lane two contiguous 16B chunks at kb and kb+16.
__device__ __forceinline__ v16bf load_a_frag(const __bf16* A, int lda, int lane) {
  FragU f;
  const int m  = lane & 15;
  const int kb = (lane >> 4) << 3;                 // 0 or 8
  const __bf16* p = A + m * lda + kb;
  f.q[0] = *reinterpret_cast<const uint4*>(p);
  f.q[1] = *reinterpret_cast<const uint4*>(p + 16);
  return f.v;
}

// B fragment 32x16 (KxN); source stored as N x K row-major (Bt[n*ldb + k]).
// Lane L: N = L&15 ; VGPR i holds K pair: K = (L>=16?8:0) + (i&3)*2 + (i>>2)*16
// => per lane two contiguous 16B chunks at kb and kb+16.
__device__ __forceinline__ v16bf load_b_frag(const __bf16* Bt, int ldb, int lane) {
  FragU f;
  const int n  = lane & 15;
  const int kb = (lane >> 4) << 3;                 // 0 or 8
  const __bf16* p = Bt + n * ldb + kb;
  f.q[0] = *reinterpret_cast<const uint4*>(p);
  f.q[1] = *reinterpret_cast<const uint4*>(p + 16);
  return f.v;
}

__device__ __forceinline__ v8f wmma_bf16(v16bf a, v16bf b, v8f c) {
  return __builtin_amdgcn_wmma_f32_16x16x32_bf16(
      /*neg_a=*/false, a, /*neg_b=*/false, b,
      /*c_mod=*/(short)0, c, /*reuse_a=*/false, /*reuse_b=*/false);
}

// symmetric per-tensor int8 quant/dequant (round-to-nearest-even like jnp.round)
__device__ __forceinline__ float quantv(float x, float s) {
  float q = rintf(x / s);
  q = fminf(fmaxf(q, -128.f), 127.f);
  return q * s;
}

// ---------------------------------------------------------------------------
// elementwise f32 -> bf16
// ---------------------------------------------------------------------------
__global__ void f32_to_bf16_kernel(const float* __restrict__ in,
                                   __bf16* __restrict__ out, int n) {
  int i = blockIdx.x * blockDim.x + threadIdx.x;
  if (i < n) out[i] = (__bf16)in[i];
}

// ---------------------------------------------------------------------------
// bf16 WMMA GEMM:  C[m,n] = sum_k A[m,k] * Bt[n,k] + bias[n]
// MODE 0: packed-QKV epilogue -> scatter q[B,H,S,D], k[B,H,S,D], vT[B,H,D,S] bf16
// MODE 1: plain f32 output [M,E]
// Block = 4 waves; wave computes 32(M) x 64(N); block tile 64 x 128.
// ---------------------------------------------------------------------------
template <int MODE>
__global__ void gemm_kernel(const __bf16* __restrict__ A,
                            const __bf16* __restrict__ Bt,
                            const float* __restrict__ bias,
                            __bf16* __restrict__ qout,
                            __bf16* __restrict__ kout,
                            __bf16* __restrict__ vtout,
                            float* __restrict__ fout) {
  const int lane = threadIdx.x & 31;
  const int wave = threadIdx.x >> 5;
  const int m0 = blockIdx.y * 64 + (wave >> 1) * 32;
  const int n0 = blockIdx.x * 128 + (wave & 1) * 64;
  constexpr int K = E;

  v8f acc[2][4];
#pragma unroll
  for (int i = 0; i < 2; ++i)
#pragma unroll
    for (int j = 0; j < 4; ++j) acc[i][j] = zero8();

  for (int k0 = 0; k0 < K; k0 += 32) {
    v16bf a0 = load_a_frag(A + (size_t)(m0 + 0)  * K + k0, K, lane);
    v16bf a1 = load_a_frag(A + (size_t)(m0 + 16) * K + k0, K, lane);
    v16bf bf[4];
#pragma unroll
    for (int j = 0; j < 4; ++j)
      bf[j] = load_b_frag(Bt + (size_t)(n0 + 16 * j) * K + k0, K, lane);
#pragma unroll
    for (int j = 0; j < 4; ++j) {
      acc[0][j] = wmma_bf16(a0, bf[j], acc[0][j]);
      acc[1][j] = wmma_bf16(a1, bf[j], acc[1][j]);
    }
  }

  const int nlane = lane & 15;
  const int mhalf = (lane >> 4) << 3;
#pragma unroll
  for (int i = 0; i < 2; ++i) {
#pragma unroll
    for (int j = 0; j < 4; ++j) {
      const int ncol = n0 + 16 * j + nlane;
      const float bv = bias[ncol];
#pragma unroll
      for (int r = 0; r < 8; ++r) {
        const int mrow = m0 + 16 * i + mhalf + r;
        const float val = acc[i][j][r] + bv;
        if (MODE == 0) {
          const int bidx = mrow >> 11;          // / S
          const int s    = mrow & (S - 1);
          const int sel  = ncol >> 10;          // / E : 0=q 1=k 2=v
          const int e    = ncol & (E - 1);
          const int h    = e >> 6;
          const int d    = e & (D - 1);
          const int bh   = bidx * H + h;
          if (sel == 0)      qout [((size_t)bh * S + s) * D + d] = (__bf16)val;
          else if (sel == 1) kout [((size_t)bh * S + s) * D + d] = (__bf16)val;
          else               vtout[((size_t)bh * D + d) * S + s] = (__bf16)val;
        } else {
          fout[(size_t)mrow * E + ncol] = val;
        }
      }
    }
  }
}

// ---------------------------------------------------------------------------
// Attention pass 1: scores = q.k^T/8 via WMMA; per-row max + global abs-max.
// One wave (32-thread block) per 16-row strip of one (b,h).
// ---------------------------------------------------------------------------
__global__ void pass_stats_kernel(const __bf16* __restrict__ qb,
                                  const __bf16* __restrict__ kb,
                                  float* __restrict__ rowmax,
                                  unsigned* __restrict__ scal) {
  const int lane  = threadIdx.x;
  const int strip = blockIdx.x;                   // NBH * (S/16)
  const int bh = strip >> 7;                      // / 128
  const int m0 = (strip & 127) << 4;
  const __bf16* q = qb + (size_t)bh * S * D;
  const __bf16* k = kb + (size_t)bh * S * D;

  const v16bf a0 = load_a_frag(q + m0 * D,      D, lane);
  const v16bf a1 = load_a_frag(q + m0 * D + 32, D, lane);

  float rmax[8];
#pragma unroll
  for (int r = 0; r < 8; ++r) rmax[r] = -3.4e38f;
  float amax = 0.f;

  for (int nt = 0; nt < S / 16; ++nt) {
    const __bf16* kt = k + (size_t)nt * 16 * D;
    v16bf b0 = load_b_frag(kt,      D, lane);
    v16bf b1 = load_b_frag(kt + 32, D, lane);
    v8f c = zero8();
    c = wmma_bf16(a0, b0, c);
    c = wmma_bf16(a1, b1, c);
#pragma unroll
    for (int r = 0; r < 8; ++r) {
      const float sv = c[r] * SCALE_QK;
      rmax[r] = fmaxf(rmax[r], sv);
      amax = fmaxf(amax, fabsf(sv));
    }
  }
#pragma unroll
  for (int mask = 1; mask <= 8; mask <<= 1) {
#pragma unroll
    for (int r = 0; r < 8; ++r)
      rmax[r] = fmaxf(rmax[r], __shfl_xor(rmax[r], mask, 32));
    amax = fmaxf(amax, __shfl_xor(amax, mask, 32));
  }
  amax = fmaxf(amax, __shfl_xor(amax, 16, 32));
  if ((lane & 15) == 0) {
    const int mb = m0 + ((lane >> 4) << 3);
#pragma unroll
    for (int r = 0; r < 8; ++r)
      rowmax[(size_t)bh * S + mb + r] = rmax[r];
  }
  if (lane == 0) atomicMax(&scal[0], __float_as_uint(amax));   // amax >= 0
}

// ---------------------------------------------------------------------------
// Attention pass 2: softmax denominators over quantized scores; global min denom.
// ---------------------------------------------------------------------------
__global__ void pass_denom_kernel(const __bf16* __restrict__ qb,
                                  const __bf16* __restrict__ kb,
                                  const float* __restrict__ rowmax,
                                  float* __restrict__ denom,
                                  unsigned* __restrict__ scal) {
  const int lane  = threadIdx.x;
  const int strip = blockIdx.x;
  const int bh = strip >> 7;
  const int m0 = (strip & 127) << 4;
  const float scale = ((const float*)scal)[1];

  const __bf16* q = qb + (size_t)bh * S * D;
  const __bf16* k = kb + (size_t)bh * S * D;
  const v16bf a0 = load_a_frag(q + m0 * D,      D, lane);
  const v16bf a1 = load_a_frag(q + m0 * D + 32, D, lane);

  const int mhalf = (lane >> 4) << 3;
  float qmax[8], sum[8];
#pragma unroll
  for (int r = 0; r < 8; ++r) {
    qmax[r] = quantv(rowmax[(size_t)bh * S + m0 + mhalf + r], scale);
    sum[r]  = 0.f;
  }

  for (int nt = 0; nt < S / 16; ++nt) {
    const __bf16* kt = k + (size_t)nt * 16 * D;
    v16bf b0 = load_b_frag(kt,      D, lane);
    v16bf b1 = load_b_frag(kt + 32, D, lane);
    v8f c = zero8();
    c = wmma_bf16(a0, b0, c);
    c = wmma_bf16(a1, b1, c);
#pragma unroll
    for (int r = 0; r < 8; ++r) {
      const float qv = quantv(c[r] * SCALE_QK, scale);
      sum[r] += __expf(qv - qmax[r]);
    }
  }
#pragma unroll
  for (int mask = 1; mask <= 8; mask <<= 1)
#pragma unroll
    for (int r = 0; r < 8; ++r)
      sum[r] += __shfl_xor(sum[r], mask, 32);

  float dmin = 3.4e38f;
  if ((lane & 15) == 0) {
#pragma unroll
    for (int r = 0; r < 8; ++r) {
      denom[(size_t)bh * S + m0 + mhalf + r] = sum[r];
      dmin = fminf(dmin, sum[r]);
    }
  }
  dmin = fminf(dmin, __shfl_xor(dmin, 16, 32));
  if (lane == 0) atomicMin(&scal[2], __float_as_uint(dmin));   // denom >= 1 > 0
}

// ---------------------------------------------------------------------------
// Attention pass 3 (fused): recompute score tiles, quantize -> softmax ->
// quantize probs -> bf16, layout-shuffle 16x32 probs tile through LDS into the
// A-fragment layout, WMMA against v^T; ctx written as [B,S,H*D] bf16.
// One wave per 16-row strip; single-wave workgroup => cheap barriers.
// ---------------------------------------------------------------------------
__global__ void pass_ctx_kernel(const __bf16* __restrict__ qb,
                                const __bf16* __restrict__ kb,
                                const __bf16* __restrict__ vtb,
                                const float* __restrict__ rowmax,
                                const float* __restrict__ denom,
                                const unsigned* __restrict__ scal,
                                __bf16* __restrict__ ctxb) {
  __shared__ __bf16 ptile[16 * 32];
  const int lane  = threadIdx.x;
  const int strip = blockIdx.x;
  const int bh = strip >> 7;
  const int m0 = (strip & 127) << 4;
  const int bidx = bh >> 4;                       // / H
  const int h    = bh & (H - 1);
  const float* fs = (const float*)scal;
  const float scale  = fs[1];
  const float scale2 = fs[3];

  const __bf16* q  = qb  + (size_t)bh * S * D;
  const __bf16* k  = kb  + (size_t)bh * S * D;
  const __bf16* vt = vtb + (size_t)bh * D * S;

  const v16bf a0 = load_a_frag(q + m0 * D,      D, lane);
  const v16bf a1 = load_a_frag(q + m0 * D + 32, D, lane);

  const int nlane = lane & 15;
  const int mhalf = (lane >> 4) << 3;
  float qmax[8], rinv[8];
#pragma unroll
  for (int r = 0; r < 8; ++r) {
    const size_t row = (size_t)bh * S + m0 + mhalf + r;
    qmax[r] = quantv(rowmax[row], scale);
    rinv[r] = 1.f / denom[row];
  }

  v8f ctx[4];
#pragma unroll
  for (int j = 0; j < 4; ++j) ctx[j] = zero8();

  for (int n0 = 0; n0 < S; n0 += 32) {
#pragma unroll
    for (int t = 0; t < 2; ++t) {
      const __bf16* kt = k + (size_t)(n0 + 16 * t) * D;
      v16bf b0 = load_b_frag(kt,      D, lane);
      v16bf b1 = load_b_frag(kt + 32, D, lane);
      v8f c = zero8();
      c = wmma_bf16(a0, b0, c);
      c = wmma_bf16(a1, b1, c);
#pragma unroll
      for (int r = 0; r < 8; ++r) {
        const float qv = quantv(c[r] * SCALE_QK, scale);
        const float p  = __expf(qv - qmax[r]) * rinv[r];
        const float pq = quantv(p, scale2);
        ptile[(mhalf + r) * 32 + 16 * t + nlane] = (__bf16)pq;
      }
    }
    __syncthreads();                               // publish probs tile
    v16bf pa = load_a_frag(ptile, 32, lane);       // ds_load_b128 x2
    __syncthreads();                               // safe to overwrite next iter
#pragma unroll
    for (int j = 0; j < 4; ++j) {
      v16bf vb = load_b_frag(vt + (size_t)(16 * j) * S + n0, S, lane);
      ctx[j] = wmma_bf16(pa, vb, ctx[j]);
    }
  }

#pragma unroll
  for (int j = 0; j < 4; ++j)
#pragma unroll
    for (int r = 0; r < 8; ++r) {
      const int srow = m0 + mhalf + r;
      const int g    = bidx * S + srow;
      const int col  = h * D + 16 * j + nlane;
      ctxb[(size_t)g * E + col] = (__bf16)ctx[j][r];
    }
}

// ---------------------------------------------------------------------------
// scalar glue kernels
// ---------------------------------------------------------------------------
__global__ void init_scalars_kernel(unsigned* scal) {
  scal[0] = 0u;                // abs-max bits
  scal[2] = 0x7F7FFFFFu;       // min-denom bits (FLT_MAX)
}
__global__ void compute_scale_kernel(unsigned* scal) {
  const float am = __uint_as_float(scal[0]);
  ((float*)scal)[1] = fmaxf(am / 127.f, 1e-8f);
}
__global__ void compute_scale2_kernel(unsigned* scal) {
  const float dm = __uint_as_float(scal[2]);
  ((float*)scal)[3] = fmaxf((1.f / dm) / 127.f, 1e-8f);   // max prob = 1/min denom
}

// ---------------------------------------------------------------------------
// host launch
// ---------------------------------------------------------------------------
extern "C" void kernel_launch(void* const* d_in, const int* in_sizes, int n_in,
                              void* d_out, int out_size, void* d_ws, size_t ws_size,
                              hipStream_t stream) {
  const float* x    = (const float*)d_in[0];
  const float* wqkv = (const float*)d_in[1];
  const float* bqkv = (const float*)d_in[2];
  const float* wo   = (const float*)d_in[3];
  const float* bo   = (const float*)d_in[4];

  char* p = (char*)d_ws;
  auto alloc = [&](size_t bytes) {
    char* r = p;
    p += (bytes + 255) & ~(size_t)255;
    return (void*)r;
  };
  __bf16*   xb     = (__bf16*)  alloc((size_t)M * E * 2);
  __bf16*   wqkvb  = (__bf16*)  alloc((size_t)N3 * E * 2);
  __bf16*   wob    = (__bf16*)  alloc((size_t)E * E * 2);
  __bf16*   qbuf   = (__bf16*)  alloc((size_t)NBH * S * D * 2);
  __bf16*   kbuf   = (__bf16*)  alloc((size_t)NBH * S * D * 2);
  __bf16*   vtbuf  = (__bf16*)  alloc((size_t)NBH * D * S * 2);
  __bf16*   ctxb   = (__bf16*)  alloc((size_t)M * E * 2);
  float*    ybuf   = (float*)   alloc((size_t)M * E * 4);
  float*    rowmax = (float*)   alloc((size_t)NBH * S * 4);
  float*    denom  = (float*)   alloc((size_t)NBH * S * 4);
  unsigned* scal   = (unsigned*)alloc(64);

  auto cvt = [&](const float* src, __bf16* dst, int n) {
    f32_to_bf16_kernel<<<(n + 255) / 256, 256, 0, stream>>>(src, dst, n);
  };
  cvt(wqkv, wqkvb, N3 * E);    // shared weights: convert once
  cvt(wo, wob, E * E);

  const int nstrips = NBH * (S / 16);   // 4096 single-wave workgroups

  for (int layer = 0; layer < 2; ++layer) {
    const float* src = (layer == 0) ? x : ybuf;
    cvt(src, xb, M * E);
    gemm_kernel<0><<<dim3(N3 / 128, M / 64), 128, 0, stream>>>(
        xb, wqkvb, bqkv, qbuf, kbuf, vtbuf, nullptr);
    init_scalars_kernel<<<1, 1, 0, stream>>>(scal);
    pass_stats_kernel<<<nstrips, 32, 0, stream>>>(qbuf, kbuf, rowmax, scal);
    compute_scale_kernel<<<1, 1, 0, stream>>>(scal);
    pass_denom_kernel<<<nstrips, 32, 0, stream>>>(qbuf, kbuf, rowmax, denom, scal);
    compute_scale2_kernel<<<1, 1, 0, stream>>>(scal);
    pass_ctx_kernel<<<nstrips, 32, 0, stream>>>(qbuf, kbuf, vtbuf, rowmax, denom,
                                                scal, ctxb);
    float* dst = (layer == 1) ? (float*)d_out : ybuf;
    gemm_kernel<1><<<dim3(E / 128, M / 64), 128, 0, stream>>>(
        ctxb, wob, bo, nullptr, nullptr, nullptr, dst);
  }
  (void)in_sizes; (void)n_in; (void)out_size; (void)ws_size;
}
